// DeltaNet_71356586656243
// MI455X (gfx1250) — matrix operand order repaired and checked by reading, not compiled, not run on hardware
//
#include <hip/hip_runtime.h>

// ---------------------------------------------------------------------------
// DeltaNet forward for MI455X (gfx1250, wave32).
// Design:
//  * All operands pre-converted f32->bf16 once (~40MB pass, ~2us at 23.3TB/s)
//    so every GEMM reads bf16 directly.
//  * GEMMs (~30 GFLOP total) use v_wmma_f32_16x16x32_bf16 with f32 accum.
//    128x64 macro-tile, 8 waves (4x2), 2x2 fragments/wave, K-step 32,
//    double-buffered LDS. A tile staged with GLOBAL_LOAD_ASYNC_TO_LDS_B128
//    (ASYNCcnt, no VGPR round-trip, overlapped with WMMA math); B tile staged
//    with uint4 loads + transpose-store so fragments are ds_load_b128 reads.
//  * Sequential delta-rule scan (16 head chains x 4096 steps, 64x64 f32
//    state) is latency-bound VALU: one block/head, state column per thread
//    in registers, fully unrolled FMAs.
// ---------------------------------------------------------------------------

#define T_LEN  2048
#define D_DIM  1024
#define H_NUM  16
#define HD_DIM 64
#define NH_SUB 2

typedef __attribute__((ext_vector_type(16))) __bf16 v16bf;
typedef __attribute__((ext_vector_type(8)))  __bf16 v8bf;
typedef __attribute__((ext_vector_type(4)))  __bf16 v4bf;
typedef __attribute__((ext_vector_type(8)))  float  v8f;

#define BM 128
#define BN 64
#define KT 32
#define KROW (KT + 8)   // LDS row pitch in bf16: 40 elems = 80B (16B-aligned, banks spread)

// ---------------------------------------------------------------------------
// f32 -> bf16 conversion (vectorized: float4 -> 4x bf16).
// ---------------------------------------------------------------------------
__global__ __launch_bounds__(256) void cvt_f32_bf16(const float* __restrict__ in,
                                                    __bf16* __restrict__ out, int n4)
{
    int i = blockIdx.x * blockDim.x + threadIdx.x;
    if (i >= n4) return;
    float4 f = ((const float4*)in)[i];
    v4bf o = {(__bf16)f.x, (__bf16)f.y, (__bf16)f.z, (__bf16)f.w};
    ((v4bf*)out)[i] = o;
}

// ---------------------------------------------------------------------------
// Tiled GEMM: C[M,N] = A[M,K] * B[K,N], bf16 in / f32 out.
// N, K compile-time so all addressing folds to immediate offsets.
// ---------------------------------------------------------------------------
template <int N, int K>
__global__ __launch_bounds__(256) void gemm_wmma_bf16(
    const __bf16* __restrict__ A, const __bf16* __restrict__ B,
    float* __restrict__ C)
{
    __shared__ __bf16 As[2][BM][KROW];   // 2 x 10 KB
    __shared__ __bf16 Bs[2][BN][KROW];   // 2 x  5 KB  (transposed: [n][k])

    const int tid   = threadIdx.x;
    const int lane  = tid & 31;
    const int wave  = tid >> 5;
    const int waveM = wave >> 1;         // 0..3
    const int waveN = wave & 1;          // 0..1
    const int half  = lane >> 4;         // wave32 half
    const int l15   = lane & 15;
    const int m0    = blockIdx.x * BM;
    const int n0    = blockIdx.y * BN;

    // Staging slot mapping (16B = 8 bf16 granules):
    // A tile 128x32: 512 granules, 2 per thread (rows ar, ar+64).
    const int ar = tid >> 2;             // 0..63
    const int ac = (tid & 3) * 8;        // 0,8,16,24
    // B tile 32x64: 256 granules, 1 per thread.
    const int br = tid >> 3;             // 0..31
    const int bc = (tid & 7) * 8;        // 0..56

    // --- A tile: async DMA global -> LDS (GLOBAL_LOAD_ASYNC_TO_LDS_B128) ---
    auto async_load_A = [&](int k0, int buf) {
        const __bf16* g0 = A + (size_t)(m0 + ar) * K + k0 + ac;
        const __bf16* g1 = g0 + (size_t)64 * K;
        // Low 32 bits of a flat LDS pointer == wave-relative LDS offset.
        unsigned l0 = (unsigned)(size_t)&As[buf][ar][ac];
        unsigned l1 = (unsigned)(size_t)&As[buf][ar + 64][ac];
        asm volatile("global_load_async_to_lds_b128 %0, %1, off"
                     :: "v"(l0), "v"(g0) : "memory");
        asm volatile("global_load_async_to_lds_b128 %0, %1, off"
                     :: "v"(l1), "v"(g1) : "memory");
    };

    // --- B tile: vector load + transpose store (fragment wants [n][k]) ---
    uint4 bR;
    auto load_B = [&](int k0) {
        if (n0 + bc + 8 <= N) {
            bR = *(const uint4*)(B + (size_t)(k0 + br) * N + n0 + bc);
        } else {
            bR = make_uint4(0u, 0u, 0u, 0u);
        }
    };
    auto store_B = [&](int buf) {
        const __bf16* pb = (const __bf16*)&bR;
        #pragma unroll
        for (int j = 0; j < 8; ++j) Bs[buf][bc + j][br] = pb[j];  // transpose
    };

    v8f acc[2][2] = {};

    async_load_A(0, 0);
    load_B(0);
    store_B(0);
    asm volatile("s_wait_asynccnt 0x0" ::: "memory");
    __syncthreads();

    const int NT = K / KT;
    for (int it = 0; it < NT; ++it) {
        const int buf = it & 1;
        if (it + 1 < NT) {
            async_load_A((it + 1) * KT, 1 - buf);   // DMA next A tile
            load_B((it + 1) * KT);                  // next B tile -> regs
        }
        if (it + 2 < NT)
            __builtin_prefetch(B + (size_t)((it + 2) * KT + br) * N + n0 + bc, 0, 1);

        // Fragments: contiguous k order, lane half h holds K = 16h..16h+15 for both A and B.
        v16bf af[2], bfv[2];
        #pragma unroll
        for (int s = 0; s < 2; ++s) {
            const __bf16* ap = &As[buf][waveM * 32 + s * 16 + l15][16 * half];
            v8bf lo = *(const v8bf*)ap;          // K = 16h .. 16h+7
            v8bf hi = *(const v8bf*)(ap + 8);    // K = 16h+8 .. 16h+15
            af[s] = __builtin_shufflevector(lo, hi,
                     0, 1, 2, 3, 4, 5, 6, 7, 8, 9, 10, 11, 12, 13, 14, 15);
            bfv[s] = *(const v16bf*)&Bs[buf][waveN * 32 + s * 16 + l15][16 * half];
        }

        #pragma unroll
        for (int mi = 0; mi < 2; ++mi)
            #pragma unroll
            for (int ni = 0; ni < 2; ++ni)
                acc[mi][ni] = __builtin_amdgcn_wmma_f32_16x16x32_bf16(
                    false, af[mi], false, bfv[ni],
                    (short)0, acc[mi][ni], false, false);

        if (it + 1 < NT) store_B(1 - buf);
        asm volatile("s_wait_asynccnt 0x0" ::: "memory");  // next A tile landed
        __syncthreads();
    }

    // Store: C/D f32 16x16 layout -> m = r + 8*half, n = lane&15.
    #pragma unroll
    for (int mi = 0; mi < 2; ++mi)
        #pragma unroll
        for (int ni = 0; ni < 2; ++ni) {
            int n = n0 + waveN * 32 + ni * 16 + l15;
            if (n < N) {
                #pragma unroll
                for (int r = 0; r < 8; ++r) {
                    int m = m0 + waveM * 32 + mi * 16 + r + 8 * half;
                    C[(size_t)m * N + n] = acc[mi][ni][r];
                }
            }
        }
}

// ---------------------------------------------------------------------------
// Row-wise L2 norm (rows of 64), optional scale. One block per row.
// ---------------------------------------------------------------------------
__global__ __launch_bounds__(64) void l2norm_kernel(float* __restrict__ data,
                                                    float scale)
{
    const int row = blockIdx.x;
    const int tid = threadIdx.x;
    __shared__ float red[64];
    float v = data[(size_t)row * 64 + tid];
    red[tid] = v * v;
    __syncthreads();
    for (int s = 32; s > 0; s >>= 1) {
        if (tid < s) red[tid] += red[tid + s];
        __syncthreads();
    }
    float r = rsqrtf(red[0] + 1e-6f) * scale;
    data[(size_t)row * 64 + tid] = v * r;
}

// ---------------------------------------------------------------------------
// beta = 2*sigmoid(braw); g = -exp(A_log)*softplus(araw+dt_bias) at sub-step
// 0, zero at sub-step 1.  Index space: T*NH*H.
// ---------------------------------------------------------------------------
__global__ void betag_kernel(const float* __restrict__ braw,
                             const float* __restrict__ araw,
                             const float* __restrict__ A_log,
                             const float* __restrict__ dt_bias,
                             float* __restrict__ beta,
                             float* __restrict__ g)
{
    int idx = blockIdx.x * blockDim.x + threadIdx.x;
    const int total = T_LEN * NH_SUB * H_NUM;
    if (idx >= total) return;
    int h  = idx % H_NUM;
    int nh = (idx / H_NUM) % NH_SUB;
    int t  = idx / (H_NUM * NH_SUB);

    float b = braw[idx]; // [T][NH][H] flat == idx
    beta[idx] = 2.0f / (1.0f + __expf(-b));

    float gv = 0.0f;
    if (nh == 0) {
        float a = araw[(size_t)t * H_NUM + h] + dt_bias[h];
        float sp = (a > 20.0f) ? a : log1pf(__expf(a));
        gv = -__expf(A_log[h]) * sp;
    }
    g[idx] = gv;
}

// ---------------------------------------------------------------------------
// Sequential delta-rule scan. One block per head (16 blocks, 64 threads).
// Thread tid owns state column S[:, tid] (64 f32 registers).
// per step: S *= exp(g); pred = k.S; dv = (v - pred)*beta; S += k (x) dv;
// on sub-step 1: o = q.S
// ---------------------------------------------------------------------------
__global__ __launch_bounds__(64) void scan_kernel(
    const float* __restrict__ q,     // [T,H,64]  (l2-normed * HD^-0.5)
    const float* __restrict__ k,     // [T*NH,H,64] (l2-normed)
    const float* __restrict__ v,     // [T*NH,H,64]
    const float* __restrict__ beta,  // [T*NH,H]
    const float* __restrict__ g,     // [T*NH,H]
    float* __restrict__ o)           // [T,H,64]
{
    const int h   = blockIdx.x;
    const int tid = threadIdx.x;

    __shared__ float ks[64], vs[64], qs[64], sc[2];

    float S[64];
    #pragma unroll
    for (int j = 0; j < 64; ++j) S[j] = 0.0f;

    for (int t = 0; t < T_LEN * NH_SUB; ++t) {
        const size_t base = ((size_t)t * H_NUM + h) * 64;
        ks[tid] = k[base + tid];
        vs[tid] = v[base + tid];
        if (t & 1) qs[tid] = q[(((size_t)(t >> 1)) * H_NUM + h) * 64 + tid];
        if (tid == 0) {
            sc[0] = beta[(size_t)t * H_NUM + h];
            sc[1] = __expf(g[(size_t)t * H_NUM + h]);
        }
        __syncthreads();

        const float bt = sc[0], eg = sc[1];
        float pred = 0.0f;
        #pragma unroll
        for (int j = 0; j < 64; ++j) {
            S[j] *= eg;
            pred = fmaf(ks[j], S[j], pred);
        }
        const float dv = (vs[tid] - pred) * bt;
        #pragma unroll
        for (int j = 0; j < 64; ++j) S[j] = fmaf(ks[j], dv, S[j]);

        if (t & 1) {
            float accum = 0.0f;
            #pragma unroll
            for (int j = 0; j < 64; ++j) accum = fmaf(qs[j], S[j], accum);
            o[(((size_t)(t >> 1)) * H_NUM + h) * 64 + tid] = accum;
        }
        __syncthreads();
    }
}

// ---------------------------------------------------------------------------
// Gated RMSNorm + swish gate. One block per (t,h) row of 64.
// ---------------------------------------------------------------------------
__global__ __launch_bounds__(64) void postnorm_kernel(
    const float* __restrict__ o_scan, const float* __restrict__ gate,
    const float* __restrict__ norm_w, float* __restrict__ o_out)
{
    const int row = blockIdx.x;
    const int tid = threadIdx.x;
    __shared__ float red[64];
    float v = o_scan[(size_t)row * 64 + tid];
    red[tid] = v * v;
    __syncthreads();
    for (int s = 32; s > 0; s >>= 1) {
        if (tid < s) red[tid] += red[tid + s];
        __syncthreads();
    }
    float r  = rsqrtf(red[0] * (1.0f / 64.0f) + 1e-5f);
    float on = v * r * norm_w[tid];
    float gv = gate[(size_t)row * 64 + tid];
    float sw = gv / (1.0f + __expf(-gv));
    o_out[(size_t)row * 64 + tid] = on * sw;
}

// ---------------------------------------------------------------------------
extern "C" void kernel_launch(void* const* d_in, const int* in_sizes, int n_in,
                              void* d_out, int out_size, void* d_ws, size_t ws_size,
                              hipStream_t stream)
{
    const float* x        = (const float*)d_in[0];
    const float* Wq       = (const float*)d_in[1];
    const float* Wk       = (const float*)d_in[2];
    const float* Wv       = (const float*)d_in[3];
    const float* Wb       = (const float*)d_in[4];
    const float* Wa       = (const float*)d_in[5];
    const float* A_log    = (const float*)d_in[6];
    const float* dt_bias  = (const float*)d_in[7];
    const float* Wg       = (const float*)d_in[8];
    const float* norm_w   = (const float*)d_in[9];
    const float* Wo       = (const float*)d_in[10];
    float* out            = (float*)d_out;

    // ---- workspace layout ----
    const size_t QSZ = (size_t)T_LEN * H_NUM * HD_DIM;           // 2M
    const size_t KSZ = (size_t)T_LEN * NH_SUB * H_NUM * HD_DIM;  // 4M
    const size_t SSZ = (size_t)T_LEN * NH_SUB * H_NUM;           // 64K
    const size_t XSZ = (size_t)T_LEN * D_DIM;                    // 2M
    const size_t WSZ = (size_t)D_DIM * D_DIM;                    // 1M

    float* f32ws = (float*)d_ws;
    float* qb   = f32ws;
    float* kb   = qb + QSZ;
    float* vb   = kb + KSZ;
    float* gate = vb + KSZ;
    float* braw = gate + QSZ;
    float* araw = braw + SSZ;
    float* beta = araw + (size_t)T_LEN * H_NUM;
    float* gdec = beta + SSZ;
    float* oscn = gdec + SSZ;
    float* onrm = oscn + QSZ;
    float* f32end = onrm + QSZ;

    __bf16* bfws  = (__bf16*)f32end;
    __bf16* xb    = bfws;
    __bf16* wqb   = xb  + XSZ;
    __bf16* wkb   = wqb + WSZ;
    __bf16* wvb   = wkb + 2 * WSZ;
    __bf16* wbb   = wvb + 2 * WSZ;
    __bf16* wab   = wbb + (size_t)D_DIM * NH_SUB * H_NUM;
    __bf16* wgb   = wab + (size_t)D_DIM * H_NUM;
    __bf16* wob   = wgb + WSZ;
    __bf16* onrmb = wob + WSZ;

    const dim3 blk256(256);
    auto cvt = [&](const float* src, __bf16* dst, size_t n) {
        int n4 = (int)(n / 4);
        cvt_f32_bf16<<<dim3((n4 + 255) / 256), blk256, 0, stream>>>(src, dst, n4);
    };

    // ---- f32 -> bf16 operand conversion (single pass) ----
    cvt(x,  xb,  XSZ);
    cvt(Wq, wqb, WSZ);
    cvt(Wk, wkb, 2 * WSZ);
    cvt(Wv, wvb, 2 * WSZ);
    cvt(Wb, wbb, (size_t)D_DIM * NH_SUB * H_NUM);
    cvt(Wa, wab, (size_t)D_DIM * H_NUM);
    cvt(Wg, wgb, WSZ);
    cvt(Wo, wob, WSZ);

    auto grid = [](int N) { return dim3(T_LEN / BM, (N + BN - 1) / BN); };

    // ---- projections (bf16 WMMA GEMMs) ----
    gemm_wmma_bf16<1024, 1024><<<grid(1024), blk256, 0, stream>>>(xb, wqb, qb);
    gemm_wmma_bf16<2048, 1024><<<grid(2048), blk256, 0, stream>>>(xb, wkb, kb);
    gemm_wmma_bf16<2048, 1024><<<grid(2048), blk256, 0, stream>>>(xb, wvb, vb);
    gemm_wmma_bf16<  32, 1024><<<grid(  32), blk256, 0, stream>>>(xb, wbb, braw);
    gemm_wmma_bf16<  16, 1024><<<grid(  16), blk256, 0, stream>>>(xb, wab, araw);
    gemm_wmma_bf16<1024, 1024><<<grid(1024), blk256, 0, stream>>>(xb, wgb, gate);

    // ---- q/k normalization ----
    l2norm_kernel<<<dim3(T_LEN * H_NUM),          dim3(64), 0, stream>>>(qb, 0.125f); // HD^-0.5
    l2norm_kernel<<<dim3(T_LEN * NH_SUB * H_NUM), dim3(64), 0, stream>>>(kb, 1.0f);

    // ---- beta / decay ----
    betag_kernel<<<dim3((T_LEN * NH_SUB * H_NUM + 255) / 256), blk256, 0, stream>>>(
        braw, araw, A_log, dt_bias, beta, gdec);

    // ---- sequential scan: one block per head ----
    scan_kernel<<<dim3(H_NUM), dim3(64), 0, stream>>>(qb, kb, vb, beta, gdec, oscn);

    // ---- gated RMSNorm + swish ----
    postnorm_kernel<<<dim3(T_LEN * H_NUM), dim3(64), 0, stream>>>(oscn, gate, norm_w, onrm);

    // ---- output projection ----
    cvt(onrm, onrmb, QSZ);
    gemm_wmma_bf16<1024, 1024><<<grid(1024), blk256, 0, stream>>>(onrmb, wob, out);
}
